// SSLPretrainModel_15118284882423
// MI455X (gfx1250) — matrix operand order repaired and checked by reading, not compiled, not run on hardware
//
#include <hip/hip_runtime.h>
#include <hip/hip_bf16.h>

// D-MPNN (ChemProp-style) for MI455X / gfx1250.
// GEMMs via v_wmma_f32_16x16x32_f16. LDS tiles are kept in WMMA *fragment
// order* so operand loads are 2x ds_load_b128 per fragment (no scalar u16
// shuffling). Gathers are float4-coalesced; scatters use global f32 atomics.

typedef __attribute__((ext_vector_type(16))) _Float16 v16h;
typedef __attribute__((ext_vector_type(8)))  float    v8f;
typedef __attribute__((ext_vector_type(4)))  _Float16 v4h;
typedef __attribute__((ext_vector_type(2)))  _Float16 v2h;

#define N_NODES 100000
#define N_EDGES 800000
#define A_FEAT  133
#define B_FEAT  14
#define N_GRAPH 1000

// ---------------------------------------------------------------------------
// Fragment-order LDS addressing (half-element indices).
//  A 16x32 f16 : (row,k) -> lane = row + 16*k[4], elem = k[3:0]
//  B 32x16 f16 : (k,n)   -> lane = n[3:0] + 16*k[4], elem = k[3:0]
// Per (chunk[,ntile]) a fragment is 32 lanes * 16 halfs, contiguous.
// ---------------------------------------------------------------------------
__device__ __forceinline__ int a_addr(int row, int k) {
    return ((k >> 5) * 32 + row + ((k >> 4) & 1) * 16) * 16
           + (k & 15);
}
__device__ __forceinline__ int b_addr(int k, int n, int NT) {
    return (((k >> 5) * NT + (n >> 4)) * 32 + (n & 15) + ((k >> 4) & 1) * 16) * 16
           + (k & 15);
}

// ---------------------------------------------------------------------------
__global__ void k_zero4(float* __restrict__ p, long n4) {
    long i = (long)blockIdx.x * blockDim.x + threadIdx.x;
    if (i < n4) ((float4*)p)[i] = make_float4(0.f, 0.f, 0.f, 0.f);
}

// segment sum: out[edst[e]*64 + 4c..4c+3] += ew[e]*h[e*64 + 4c..]; total = E*16
__global__ void k_seg_sum_scaled(const float* __restrict__ h,
                                 const float* __restrict__ ew,
                                 const int*   __restrict__ edst,
                                 float* __restrict__ out, long total16) {
    long i = (long)blockIdx.x * 256 + threadIdx.x;
    if (i >= total16) return;
    long e = i >> 4;
    int  c = (int)(i & 15) * 4;
    float w = ew[e];
    float4 hv = *(const float4*)&h[e * 64 + c];
    float* o = &out[(long)edst[e] * 64 + c];
    atomicAdd(o + 0, w * hv.x);
    atomicAdd(o + 1, w * hv.y);
    atomicAdd(o + 2, w * hv.z);
    atomicAdd(o + 3, w * hv.w);
}

// graph pooling: ge[n2g[v]*64 + 4c..] += nr[v*64 + 4c..]; total = N*16
__global__ void k_graph_sum(const float* __restrict__ nr,
                            const int*   __restrict__ n2g,
                            float* __restrict__ ge, long total16) {
    long i = (long)blockIdx.x * 256 + threadIdx.x;
    if (i >= total16) return;
    long v = i >> 4;
    int  c = (int)(i & 15) * 4;
    float4 hv = *(const float4*)&nr[v * 64 + c];
    float* o = &ge[(long)n2g[v] * 64 + c];
    atomicAdd(o + 0, hv.x);
    atomicAdd(o + 1, hv.y);
    atomicAdd(o + 2, hv.z);
    atomicAdd(o + 3, hv.w);
}

// ---------------------------------------------------------------------------
// Kernel 1: h0 = relu(concat(atom[esrc], efeat) @ Wi)   (E x 147)@(147 x 64)
// K padded 147 -> 160 (5 chunks). 8 waves/block, 16 rows/wave.
// ---------------------------------------------------------------------------
__global__ void __launch_bounds__(256)
k_edge_in_gemm(const float* __restrict__ atom,
               const float* __restrict__ efeat,
               const int*   __restrict__ esrc,
               const float* __restrict__ Wi,
               float* __restrict__ hout, int E) {
    __shared__ alignas(32) _Float16 sW[5 * 4 * 32 * 16];   // 20 KB, frag order
    __shared__ alignas(32) _Float16 sA[8][5 * 32 * 16];    // 40 KB, frag order
    int tid = threadIdx.x, wave = tid >> 5, lane = tid & 31;
    int half = lane >> 4, l15 = lane & 15;

    // weights -> fragment order (pairs of k)
    for (int i = tid; i < 80 * 64; i += 256) {
        int kp = i >> 6, n = i & 63, k = kp * 2;
        float w0 = (k     < 147) ? Wi[(long)k * 64 + n]       : 0.0f;
        float w1 = (k + 1 < 147) ? Wi[(long)(k + 1) * 64 + n] : 0.0f;
        v2h p = {(_Float16)w0, (_Float16)w1};
        *(v2h*)&sW[b_addr(k, n, 4)] = p;
    }
    // stage A rows: each half-wave handles a row, 4 feats per lane per pass
    long rowbase = (long)blockIdx.x * 128 + wave * 16;
    for (int rp = 0; rp < 16; rp += 2) {
        int r = rp + half;
        long e = rowbase + r; if (e >= E) e = E - 1;
        int src = esrc[e];
        for (int j4 = l15 * 4; j4 < 160; j4 += 64) {
            float v[4];
#pragma unroll
            for (int t = 0; t < 4; ++t) {
                int j = j4 + t;
                v[t] = (j < 133) ? atom[(long)src * 133 + j]
                     : (j < 147) ? efeat[e * 14 + (j - 133)] : 0.0f;
            }
            v4h p = {(_Float16)v[0], (_Float16)v[1], (_Float16)v[2], (_Float16)v[3]};
            *(v4h*)&sA[wave][a_addr(r, j4)] = p;
        }
    }
    __syncthreads();

    v8f acc[4] = {};
#pragma unroll
    for (int kc = 0; kc < 5; ++kc) {
        v16h af = *(const v16h*)&sA[wave][(kc * 32 + lane) * 16];
#pragma unroll
        for (int nt = 0; nt < 4; ++nt) {
            v16h bf = *(const v16h*)&sW[((kc * 4 + nt) * 32 + lane) * 16];
            acc[nt] = __builtin_amdgcn_wmma_f32_16x16x32_f16(
                false, af, false, bf, (short)0, acc[nt], false, false);
        }
    }
#pragma unroll
    for (int nt = 0; nt < 4; ++nt) {
        int col = nt * 16 + l15;
#pragma unroll
        for (int r = 0; r < 8; ++r) {
            long e = rowbase + r + 8 * half;
            if (e < E) {
                float v = acc[nt][r];
                hout[e * 64 + col] = v > 0.0f ? v : 0.0f;
            }
        }
    }
}

// ---------------------------------------------------------------------------
// Kernel 2 (per depth): h' = relu((M[esrc] - ew[rev]*h[rev]) @ Wm + bm)
// ---------------------------------------------------------------------------
__global__ void __launch_bounds__(256)
k_msg_gemm(const float* __restrict__ Mb,
           const float* __restrict__ hin,
           const int*   __restrict__ esrc,
           const int*   __restrict__ b2rev,
           const float* __restrict__ ew,
           const float* __restrict__ Wm,
           const float* __restrict__ bm,
           float* __restrict__ hout, int E) {
    __shared__ alignas(32) _Float16 sW[2 * 4 * 32 * 16];   // 8 KB
    __shared__ float sB[64];
    __shared__ alignas(32) _Float16 sA[8][2 * 32 * 16];    // 16 KB
    int tid = threadIdx.x, wave = tid >> 5, lane = tid & 31;
    int half = lane >> 4, l15 = lane & 15;

    for (int i = tid; i < 32 * 64; i += 256) {
        int kp = i >> 6, n = i & 63, k = kp * 2;
        v2h p = {(_Float16)Wm[(long)k * 64 + n], (_Float16)Wm[(long)(k + 1) * 64 + n]};
        *(v2h*)&sW[b_addr(k, n, 4)] = p;
    }
    if (tid < 64) sB[tid] = bm[tid];

    long rowbase = (long)blockIdx.x * 128 + wave * 16;
    for (int rp = 0; rp < 16; rp += 2) {
        int r = rp + half;
        long e = rowbase + r; if (e >= E) e = E - 1;
        int src = esrc[e];
        int rev = b2rev[e];
        float rw = ew[rev];
        int j = l15 * 4;
        float4 mv = *(const float4*)&Mb[(long)src * 64 + j];
        float4 hv = *(const float4*)&hin[(long)rev * 64 + j];
        v4h p = {(_Float16)(mv.x - rw * hv.x), (_Float16)(mv.y - rw * hv.y),
                 (_Float16)(mv.z - rw * hv.z), (_Float16)(mv.w - rw * hv.w)};
        *(v4h*)&sA[wave][a_addr(r, j)] = p;
    }
    __syncthreads();

    v8f acc[4] = {};
#pragma unroll
    for (int kc = 0; kc < 2; ++kc) {
        v16h af = *(const v16h*)&sA[wave][(kc * 32 + lane) * 16];
#pragma unroll
        for (int nt = 0; nt < 4; ++nt) {
            v16h bf = *(const v16h*)&sW[((kc * 4 + nt) * 32 + lane) * 16];
            acc[nt] = __builtin_amdgcn_wmma_f32_16x16x32_f16(
                false, af, false, bf, (short)0, acc[nt], false, false);
        }
    }
#pragma unroll
    for (int nt = 0; nt < 4; ++nt) {
        int col = nt * 16 + l15;
#pragma unroll
        for (int r = 0; r < 8; ++r) {
            long e = rowbase + r + 8 * half;
            if (e < E) {
                float v = acc[nt][r] + sB[col];
                hout[e * 64 + col] = v > 0.0f ? v : 0.0f;
            }
        }
    }
}

// ---------------------------------------------------------------------------
// Kernel 3: two-layer head  Y = relu(X@W1 + B1) @ W2 + B2
// X: R x 64, W2: 64 x OutN (padded to 144 = 9 N-tiles in LDS).
// Stage-1 output is written directly in fragment order for stage 2
// (same-wave LDS ops are in-order; no extra block barrier needed).
// ---------------------------------------------------------------------------
__global__ void __launch_bounds__(256)
k_head2(const float* __restrict__ X,
        const float* __restrict__ W1, const float* __restrict__ B1,
        const float* __restrict__ W2, const float* __restrict__ B2,
        float* __restrict__ Y, int R, int OutN) {
    __shared__ alignas(32) _Float16 sW1[2 * 4 * 32 * 16];  // 8 KB
    __shared__ alignas(32) _Float16 sW2[2 * 9 * 32 * 16];  // 18 KB
    __shared__ float sB1[64];
    __shared__ alignas(32) _Float16 sA[8][2 * 32 * 16];    // 16 KB
    __shared__ alignas(32) _Float16 sT[8][2 * 32 * 16];    // 16 KB
    int tid = threadIdx.x, wave = tid >> 5, lane = tid & 31;
    int half = lane >> 4, l15 = lane & 15;
    int ntiles2 = (OutN + 15) >> 4;

    for (int i = tid; i < 32 * 64; i += 256) {
        int kp = i >> 6, n = i & 63, k = kp * 2;
        v2h p = {(_Float16)W1[(long)k * 64 + n], (_Float16)W1[(long)(k + 1) * 64 + n]};
        *(v2h*)&sW1[b_addr(k, n, 4)] = p;
    }
    for (int i = tid; i < 32 * 144; i += 256) {
        int kp = i / 144, n = i % 144, k = kp * 2;
        float w0 = (n < OutN) ? W2[(long)k * OutN + n]       : 0.0f;
        float w1 = (n < OutN) ? W2[(long)(k + 1) * OutN + n] : 0.0f;
        v2h p = {(_Float16)w0, (_Float16)w1};
        *(v2h*)&sW2[b_addr(k, n, 9)] = p;
    }
    if (tid < 64) sB1[tid] = B1[tid];

    long rowbase = (long)blockIdx.x * 128 + wave * 16;
    for (int rp = 0; rp < 16; rp += 2) {
        int r = rp + half;
        long x = rowbase + r; if (x >= R) x = R - 1;
        int j = l15 * 4;
        float4 xv = *(const float4*)&X[x * 64 + j];
        v4h p = {(_Float16)xv.x, (_Float16)xv.y, (_Float16)xv.z, (_Float16)xv.w};
        *(v4h*)&sA[wave][a_addr(r, j)] = p;
    }
    __syncthreads();

    // stage 1: t = relu(X@W1 + B1) -> sT in fragment order
    {
        v8f acc[4] = {};
#pragma unroll
        for (int kc = 0; kc < 2; ++kc) {
            v16h af = *(const v16h*)&sA[wave][(kc * 32 + lane) * 16];
#pragma unroll
            for (int nt = 0; nt < 4; ++nt) {
                v16h bf = *(const v16h*)&sW1[((kc * 4 + nt) * 32 + lane) * 16];
                acc[nt] = __builtin_amdgcn_wmma_f32_16x16x32_f16(
                    false, af, false, bf, (short)0, acc[nt], false, false);
            }
        }
#pragma unroll
        for (int nt = 0; nt < 4; ++nt) {
            int col = nt * 16 + l15;
#pragma unroll
            for (int r = 0; r < 8; ++r) {
                int rowin = r + 8 * half;
                float v = acc[nt][r] + sB1[col];
                sT[wave][a_addr(rowin, col)] = (_Float16)(v > 0.0f ? v : 0.0f);
            }
        }
    }

    // stage 2: Y = t @ W2 + B2
    v16h af0 = *(const v16h*)&sT[wave][(0 * 32 + lane) * 16];
    v16h af1 = *(const v16h*)&sT[wave][(1 * 32 + lane) * 16];
    for (int nt2 = 0; nt2 < ntiles2; ++nt2) {
        v8f acc = {};
        v16h bf0 = *(const v16h*)&sW2[((0 * 9 + nt2) * 32 + lane) * 16];
        acc = __builtin_amdgcn_wmma_f32_16x16x32_f16(
            false, af0, false, bf0, (short)0, acc, false, false);
        v16h bf1 = *(const v16h*)&sW2[((1 * 9 + nt2) * 32 + lane) * 16];
        acc = __builtin_amdgcn_wmma_f32_16x16x32_f16(
            false, af1, false, bf1, (short)0, acc, false, false);
        int col = nt2 * 16 + l15;
        if (col < OutN) {
            float bias = B2[col];
#pragma unroll
            for (int r = 0; r < 8; ++r) {
                long row = rowbase + r + 8 * half;
                if (row < R) Y[row * OutN + col] = acc[r] + bias;
            }
        }
    }
}

// ---------------------------------------------------------------------------
// Kernel 4: tiny graph head (G=1000)
// ---------------------------------------------------------------------------
__global__ void k_graph_head(const float* __restrict__ ge,
                             const float* __restrict__ w1, const float* __restrict__ b1,
                             const float* __restrict__ w2, const float* __restrict__ b2,
                             float* __restrict__ out, int G) {
    int g = blockIdx.x * 64 + threadIdx.x;
    if (g >= G) return;
    float x[64];
#pragma unroll
    for (int j = 0; j < 64; ++j) x[j] = ge[(long)g * 64 + j];
    float s = b2[0];
    for (int j = 0; j < 64; ++j) {
        float t = b1[j];
        for (int k = 0; k < 64; ++k) t += x[k] * w1[k * 64 + j];
        s += (t > 0.0f ? t : 0.0f) * w2[j];
    }
    out[g] = s;
}

// ---------------------------------------------------------------------------
extern "C" void kernel_launch(void* const* d_in, const int* in_sizes, int n_in,
                              void* d_out, int out_size, void* d_ws, size_t ws_size,
                              hipStream_t stream) {
    const float* atom    = (const float*)d_in[0];
    const float* efeat   = (const float*)d_in[1];
    const float* ew      = (const float*)d_in[2];
    const float* Wi      = (const float*)d_in[3];
    const float* Wm_w    = (const float*)d_in[4];
    const float* Wm_b    = (const float*)d_in[5];
    const float* node_w1 = (const float*)d_in[6];
    const float* node_b1 = (const float*)d_in[7];
    const float* node_w2 = (const float*)d_in[8];
    const float* node_b2 = (const float*)d_in[9];
    const float* edge_w1 = (const float*)d_in[10];
    const float* edge_b1 = (const float*)d_in[11];
    const float* edge_w2 = (const float*)d_in[12];
    const float* edge_b2 = (const float*)d_in[13];
    const float* gr_w1   = (const float*)d_in[14];
    const float* gr_b1   = (const float*)d_in[15];
    const float* gr_w2   = (const float*)d_in[16];
    const float* gr_b2   = (const float*)d_in[17];
    const int*   esrc    = (const int*)d_in[18];
    const int*   edst    = (const int*)d_in[19];
    const int*   b2rev   = (const int*)d_in[20];
    const int*   n2g     = (const int*)d_in[21];

    const int N = N_NODES, E = N_EDGES, G = N_GRAPH;

    // workspace: hA, hB (E x 64 each), Mbuf (N x 64)
    float* hA   = (float*)d_ws;
    float* hB   = hA + (size_t)E * 64;
    float* Mbuf = hB + (size_t)E * 64;

    // d_out layout (flat, return order)
    float* pred_node    = (float*)d_out;
    float* pred_edge    = pred_node + (size_t)N * A_FEAT;
    float* pred_graph   = pred_edge + (size_t)E * B_FEAT;
    float* graph_embeds = pred_graph + G;
    float* node_repr    = graph_embeds + (size_t)G * 64;
    float* h_final      = node_repr + (size_t)N * 64;

    const unsigned gE128 = (E + 127) / 128;   // 6250
    const unsigned gN128 = (N + 127) / 128;   // 782
    const long tE16 = (long)E * 16;
    const long tN16 = (long)N * 16;
    const long zN4  = (long)N * 16;           // (N*64)/4 float4s
    const long zG4  = (long)G * 16;

    // 1) h0
    k_edge_in_gemm<<<gE128, 256, 0, stream>>>(atom, efeat, esrc, Wi, hA, E);

    // 2) DEPTH = 3 message-passing iterations (ping-pong; last -> d_out.h)
    const float* hcur = hA;
    for (int d = 0; d < 3; ++d) {
        float* hnext = (d == 0) ? hB : (d == 1) ? hA : h_final;
        k_zero4<<<(unsigned)((zN4 + 255) / 256), 256, 0, stream>>>(Mbuf, zN4);
        k_seg_sum_scaled<<<(unsigned)((tE16 + 255) / 256), 256, 0, stream>>>(
            hcur, ew, edst, Mbuf, tE16);
        k_msg_gemm<<<gE128, 256, 0, stream>>>(Mbuf, hcur, esrc, b2rev, ew,
                                              Wm_w, Wm_b, hnext, E);
        hcur = hnext;
    }

    // 3) node_repr = segment_sum(ew * h, edge_dst)
    k_zero4<<<(unsigned)((zN4 + 255) / 256), 256, 0, stream>>>(node_repr, zN4);
    k_seg_sum_scaled<<<(unsigned)((tE16 + 255) / 256), 256, 0, stream>>>(
        h_final, ew, edst, node_repr, tE16);

    // 4) graph_embeds = segment_sum(node_repr, node_to_graph)
    k_zero4<<<(unsigned)((zG4 + 255) / 256), 256, 0, stream>>>(graph_embeds, zG4);
    k_graph_sum<<<(unsigned)((tN16 + 255) / 256), 256, 0, stream>>>(
        node_repr, n2g, graph_embeds, tN16);

    // 5) heads
    k_head2<<<gN128, 256, 0, stream>>>(node_repr, node_w1, node_b1,
                                       node_w2, node_b2, pred_node, N, A_FEAT);
    k_head2<<<gE128, 256, 0, stream>>>(h_final, edge_w1, edge_b1,
                                       edge_w2, edge_b2, pred_edge, E, B_FEAT);
    k_graph_head<<<(G + 63) / 64, 64, 0, stream>>>(graph_embeds, gr_w1, gr_b1,
                                                   gr_w2, gr_b2, pred_graph, G);
}
